// SelfAttention_67688684585395
// MI455X (gfx1250) — matrix operand order repaired and checked by reading, not compiled, not run on hardware
//
#include <hip/hip_runtime.h>
#include <hip/hip_bf16.h>

// MI455X (gfx1250) windowed self-attention.
// HBM-bound (~150MB traffic, ~9 GFLOP): fused QKV (x read once), banded
// |i-j|<=64 attention, f16 WMMA / f32 accumulate, TDM double-buffered weight
// staging (tensor_load_to_lds + s_wait_tensorcnt) in the QKV kernel.

typedef __attribute__((ext_vector_type(16))) _Float16 v16h;
typedef __attribute__((ext_vector_type(8)))  float    v8f;
typedef __attribute__((ext_vector_type(4)))  unsigned int u32x4;
typedef __attribute__((ext_vector_type(8)))  int      i32x8;
typedef __attribute__((ext_vector_type(4)))  int      i32x4;

#define T_SEQ 2048
#define H_DIM 1024
#define P_DIM 64
#define WIN   64

#if defined(__gfx1250__) && __has_builtin(__builtin_amdgcn_tensor_load_to_lds)
#define USE_TDM 1
#else
#define USE_TDM 0
#endif

static __device__ __forceinline__ v8f wmma_f16(v16h a, v16h b, v8f c) {
  return __builtin_amdgcn_wmma_f32_16x16x32_f16(false, a, false, b, (short)0, c,
                                                false, false);
}

// A-fragment (16x32, f16) from an f32 row-major matrix with leading dim ld.
// lane=(m16,hi): halves 0..15 -> K = 2*hiOff+0..15
static __device__ __forceinline__ v16h load_a_f32(const float* __restrict__ base,
                                                  int ld, int m16, int hiOff) {
  const float* r = base + (size_t)m16 * ld + 2 * hiOff;
  v16h a;
#pragma unroll
  for (int h = 0; h < 8; ++h) {
    a[h]     = (_Float16)r[h];
    a[8 + h] = (_Float16)r[8 + h];
  }
  return a;
}

// B-fragment (32x16, f16) from a raw row-major f32 [32 x 64] LDS chunk.
// lane=(m16,hi): half h -> K = h + 16*hi, N = c*16 + m16.
static __device__ __forceinline__ v16h load_b_lds_f32(const float* buf, int hi,
                                                      int c, int m16) {
  const float* p = buf + (size_t)(hi * 16) * 64 + c * 16 + m16;
  v16h b;
#pragma unroll
  for (int h = 0; h < 16; ++h) b[h] = (_Float16)p[h * 64];
  return b;
}

#if USE_TDM
// Issue a TDM 2D tile load: 32 rows x 64 f32 cols from a [1024][64] row-major
// tensor into LDS at byte offset lds_off.  D# layout per CDNA5 ISA ch.8.
static __device__ __forceinline__ void tdm_load_w_chunk(unsigned lds_off,
                                                        const float* gaddr) {
  unsigned long long ga = (unsigned long long)(uintptr_t)gaddr;
  u32x4 g0;
  g0[0] = 1u;                                          // count=1, user mode
  g0[1] = lds_off;                                     // lds_addr (bytes)
  g0[2] = (unsigned)(ga & 0xffffffffu);                // global_addr[31:0]
  g0[3] = (unsigned)((ga >> 32) & 0x1ffffffu)          // global_addr[56:32]
          | (2u << 30);                                // type=2 (image)
  i32x8 g1;
  g1[0] = (2 << 16);                 // data_size = 4 bytes; no multicast/pad
  g1[1] = (64 << 16);                // tensor_dim0 = 64 (lo16 at bit 16)
  g1[2] = (1024 << 16);              // tensor_dim0 hi=0 | tensor_dim1=1024 lo16
  g1[3] = (64 << 16);                // tensor_dim1 hi=0 | tile_dim0 = 64
  g1[4] = 32;                        // tile_dim1 = 32, tile_dim2 = 0
  g1[5] = 64;                        // tensor_dim0_stride = 64 (lo32)
  g1[6] = 0;                         // stride0 hi | stride1 lo (2D: unused)
  g1[7] = 0;
  i32x4 z4 = {0, 0, 0, 0};
#if __clang_major__ >= 23
  i32x8 z8 = {0, 0, 0, 0, 0, 0, 0, 0};
  __builtin_amdgcn_tensor_load_to_lds(g0, g1, z4, z4, z8, 0);
#else
  __builtin_amdgcn_tensor_load_to_lds(g0, g1, z4, z4, 0);
#endif
}
#endif

// ---------------------------------------------------------------------------
// Kernel 1: fused QKV projection.  q,k natural [R,64]; v transposed [B][64][T].
// Weight chunks (3 x 32x64 f32) staged into double-buffered LDS via the Tensor
// Data Mover (wave 0 issues, s_wait_tensorcnt, workgroup barrier publishes),
// overlapping chunk i+1's DMA with chunk i's 12 WMMAs.
// ---------------------------------------------------------------------------
__global__ __launch_bounds__(256) void qkv_kernel(
    const float* __restrict__ x,
    const float* __restrict__ Wq, const float* __restrict__ bq,
    const float* __restrict__ Wk, const float* __restrict__ bk,
    const float* __restrict__ Wv, const float* __restrict__ bv,
    float* __restrict__ q, float* __restrict__ k, float* __restrict__ vT) {
  __shared__ __align__(16) float ldsW[2][3 * 32 * 64];  // double buffer, 48KB

  const int tid  = threadIdx.x;
  const int wave = tid >> 5;
  const int lane = tid & 31;
  const int m16  = lane & 15;
  const int hi   = lane >> 4;
  const int hiOff = hi ? 8 : 0;
  const int rowBase = (blockIdx.x * 8 + wave) * 16;
  const float* Ws[3] = {Wq, Wk, Wv};

  v8f accQ[4], accK[4], accV[4];
#pragma unroll
  for (int c = 0; c < 4; ++c)
#pragma unroll
    for (int r = 0; r < 8; ++r) { accQ[c][r] = 0.f; accK[c][r] = 0.f; accV[c][r] = 0.f; }

#if USE_TDM
  if (wave == 0) {
#pragma unroll
    for (int m = 0; m < 3; ++m)
      tdm_load_w_chunk((unsigned)(uintptr_t)&ldsW[0][m * 2048], Ws[m]);
  }
#endif

  for (int it = 0; it < 32; ++it) {
    const int k0 = it * 32;
    const float* buf = ldsW[it & 1];

#if USE_TDM
    if (wave == 0) {
      if (it + 1 < 32) {
        // Prefetch next chunk into the other buffer, then wait for current.
#pragma unroll
        for (int m = 0; m < 3; ++m)
          tdm_load_w_chunk((unsigned)(uintptr_t)&ldsW[(it + 1) & 1][m * 2048],
                           Ws[m] + (size_t)(k0 + 32) * 64);
        __builtin_amdgcn_s_wait_tensorcnt(3);  // current chunk's 3 loads done
      } else {
        __builtin_amdgcn_s_wait_tensorcnt(0);
      }
    }
#else
    // Fallback: cooperative staging (same raw layout, same consume path).
#pragma unroll
    for (int e = 0; e < 24; ++e) {
      int id = tid + e * 256;
      int m = id >> 11, rem = id & 2047, kl = rem >> 6, n = rem & 63;
      const float* W = (m == 0) ? Wq : ((m == 1) ? Wk : Wv);
      ((float*)buf)[m * 2048 + kl * 64 + n] = W[(size_t)(k0 + kl) * 64 + n];
    }
#endif
    __syncthreads();

    const float* xr = x + (size_t)rowBase * H_DIM + k0;
    __builtin_prefetch(xr + (size_t)m16 * H_DIM + 64, 0, 0);  // global_prefetch_b8
    v16h a = load_a_f32(xr, H_DIM, m16, hiOff);

#pragma unroll
    for (int c = 0; c < 4; ++c) {
      v16h bQ = load_b_lds_f32(buf + 0 * 2048, hi, c, m16);
      v16h bK = load_b_lds_f32(buf + 1 * 2048, hi, c, m16);
      v16h bV = load_b_lds_f32(buf + 2 * 2048, hi, c, m16);
      accQ[c] = wmma_f16(a, bQ, accQ[c]);
      accK[c] = wmma_f16(a, bK, accK[c]);
      accV[c] = wmma_f16(a, bV, accV[c]);
    }
    __syncthreads();
  }

  const int b = rowBase >> 11;  // rowBase / T_SEQ
#pragma unroll
  for (int c = 0; c < 4; ++c) {
#pragma unroll
    for (int r = 0; r < 8; ++r) {
      int M   = r + hi * 8;
      int gr  = rowBase + M;
      int col = c * 16 + m16;
      q[(size_t)gr * 64 + col] = accQ[c][r] + bq[col];
      k[(size_t)gr * 64 + col] = accK[c][r] + bk[col];
      vT[((size_t)(b * 64 + col)) * T_SEQ + (gr & (T_SEQ - 1))] = accV[c][r] + bv[col];
    }
  }
}

// ---------------------------------------------------------------------------
// Kernel 2: banded attention, one wave per 16-row query tile.
// Pass A: 9 col-tiles of QK^T (WMMA) + window mask + rel-pos bias -> LDS f32,
//         row-max via 16-lane butterflies.  (Padding mask is softmax-invariant
//         -> dropped; out-of-window exp underflows to exactly 0 as in ref.)
// Pass B: probabilities f16 -> LDS row-major [16][160], row-sum, then P@V (WMMA).
// ---------------------------------------------------------------------------
__global__ __launch_bounds__(32) void attn_kernel(
    const float* __restrict__ q, const float* __restrict__ k,
    const float* __restrict__ vT, const float* __restrict__ bias,
    float* __restrict__ ctx) {
  __shared__ __align__(32) float    sc[16 * 160];
  __shared__ __align__(32) _Float16 ph[16 * 160];

  const int lane = threadIdx.x;
  const int m16  = lane & 15;
  const int hi   = lane >> 4;
  const int hiOff = hi ? 8 : 0;
  const int g0 = blockIdx.x * 16;      // global query row base
  const int b  = g0 >> 11;
  const int i0 = g0 & (T_SEQ - 1);
  const float invs = 0.125f;           // 1/sqrt(P)

  const float* kb = k  + (size_t)b * T_SEQ * 64;
  const float* vb = vT + (size_t)b * 64 * T_SEQ;

  // Q A-fragments (P = 64 -> two 32-deep chunks)
  v16h aq[2];
#pragma unroll
  for (int c2 = 0; c2 < 2; ++c2)
    aq[c2] = load_a_f32(q + (size_t)g0 * 64 + c2 * 32, 64, m16, hiOff);

  float rmax[8];
#pragma unroll
  for (int r = 0; r < 8; ++r) rmax[r] = -__builtin_inff();

  // ---- Pass A: scores ----
#pragma unroll
  for (int jt = 0; jt < 9; ++jt) {
    int j0 = i0 - WIN + jt * 16;
    int j  = j0 + m16;
    int jc = j < 0 ? 0 : (j > T_SEQ - 1 ? T_SEQ - 1 : j);

    v8f s;
#pragma unroll
    for (int r = 0; r < 8; ++r) s[r] = 0.f;

#pragma unroll
    for (int c2 = 0; c2 < 2; ++c2) {
      const float* kr = kb + (size_t)jc * 64 + c2 * 32 + hi * 16;
      v16h bf;
#pragma unroll
      for (int h = 0; h < 16; ++h) bf[h] = (_Float16)kr[h];
      s = wmma_f16(aq[c2], bf, s);
    }

#pragma unroll
    for (int r = 0; r < 8; ++r) {
      int i   = i0 + r + hi * 8;
      int rel = i - j;                          // in [-79, 79]
      bool in = (j >= 0) && (j < T_SEQ) && (rel <= WIN) && (rel >= -WIN);
      float bval = bias[2048 - rel];            // index in [1969,2127] -> safe
      float val  = in ? (s[r] * invs + bval) : -__builtin_inff();
      sc[(r + hi * 8) * 160 + jt * 16 + m16] = val;
      rmax[r] = fmaxf(rmax[r], val);
    }
  }
#pragma unroll
  for (int w = 1; w < 16; w <<= 1)
#pragma unroll
    for (int r = 0; r < 8; ++r)
      rmax[r] = fmaxf(rmax[r], __shfl_xor(rmax[r], w, 16));

  // ---- Pass B: probabilities ----
  float lsum[8];
#pragma unroll
  for (int r = 0; r < 8; ++r) lsum[r] = 0.f;

  for (int e = lane; e < 256; e += 32)          // zero pad cols 144..159
    ph[(e >> 4) * 160 + 144 + (e & 15)] = (_Float16)0.f;

#pragma unroll
  for (int jt = 0; jt < 9; ++jt) {
#pragma unroll
    for (int r = 0; r < 8; ++r) {
      int M = r + hi * 8;
      float p = __expf(sc[M * 160 + jt * 16 + m16] - rmax[r]);
      lsum[r] += p;
      ph[M * 160 + jt * 16 + m16] = (_Float16)p;
    }
  }
#pragma unroll
  for (int w = 1; w < 16; w <<= 1)
#pragma unroll
    for (int r = 0; r < 8; ++r) lsum[r] += __shfl_xor(lsum[r], w, 16);

  // ---- P @ V ----
  v8f acc[4];
#pragma unroll
  for (int c = 0; c < 4; ++c)
#pragma unroll
    for (int r = 0; r < 8; ++r) acc[c][r] = 0.f;

#pragma unroll
  for (int pr = 0; pr < 5; ++pr) {              // 5 pairs cover 160 cols
    int j0 = i0 - WIN + pr * 32;
    const _Float16* prow = ph + m16 * 160 + pr * 32 + 2 * hiOff;
    v16h ap;
#pragma unroll
    for (int h = 0; h < 8; ++h) { ap[h] = prow[h]; ap[8 + h] = prow[8 + h]; }

#pragma unroll
    for (int c = 0; c < 4; ++c) {
      const float* vr = vb + (size_t)(c * 16 + m16) * T_SEQ;
      v16h bf;
#pragma unroll
      for (int h = 0; h < 16; ++h) {
        int jj = j0 + hi * 16 + h;
        jj = jj < 0 ? 0 : (jj > T_SEQ - 1 ? T_SEQ - 1 : jj);  // p==0 outside
        bf[h] = (_Float16)vr[jj];
      }
      acc[c] = wmma_f16(ap, bf, acc[c]);
    }
  }

#pragma unroll
  for (int c = 0; c < 4; ++c)
#pragma unroll
    for (int r = 0; r < 8; ++r) {
      int M = r + hi * 8;
      ctx[(size_t)(g0 + M) * 64 + c * 16 + m16] = acc[c][r] / lsum[r];
    }
}

// ---------------------------------------------------------------------------
// Kernel 3: out = ctx @ Wo + bo.  Wo column panel staged in B-frag layout
// (each lane's fragment read is two contiguous ds_load_b128).
// ---------------------------------------------------------------------------
__global__ __launch_bounds__(256) void out_kernel(
    const float* __restrict__ ctx, const float* __restrict__ Wo,
    const float* __restrict__ bo, float* __restrict__ out) {
  __shared__ __align__(32) _Float16 ldsW[2 * 4 * 32 * 16];  // [chunk][ct][lane][h]

  const int tid  = threadIdx.x;
  const int wave = tid >> 5;
  const int lane = tid & 31;
  const int m16  = lane & 15;
  const int hi   = lane >> 4;
  const int hiOff = hi ? 8 : 0;
  const int rowBase = (blockIdx.x * 8 + wave) * 16;
  const int n0 = blockIdx.y * 64;

#pragma unroll
  for (int e = 0; e < 16; ++e) {
    int id = tid + e * 256;
    int kl = id >> 6, n = id & 63;
    float w = Wo[(size_t)kl * H_DIM + n0 + n];
    int chunk = kl >> 5, klc = kl & 31;
    int ct = n >> 4, nn = n & 15;
    int ln = nn + ((klc & 16) ? 16 : 0);
    ldsW[(((chunk * 4) + ct) * 32 + ln) * 16 + (klc & 15)] = (_Float16)w;
  }
  __syncthreads();

  v8f acc[4];
#pragma unroll
  for (int c = 0; c < 4; ++c)
#pragma unroll
    for (int r = 0; r < 8; ++r) acc[c][r] = 0.f;

#pragma unroll
  for (int chunk = 0; chunk < 2; ++chunk) {
    v16h a = load_a_f32(ctx + (size_t)rowBase * 64 + chunk * 32, 64, m16, hiOff);
#pragma unroll
    for (int c = 0; c < 4; ++c) {
      v16h bf = *(const v16h*)&ldsW[((chunk * 4 + c) * 32 + lane) * 16];
      acc[c] = wmma_f16(a, bf, acc[c]);
    }
  }

#pragma unroll
  for (int c = 0; c < 4; ++c)
#pragma unroll
    for (int r = 0; r < 8; ++r) {
      int M   = r + hi * 8;
      int col = n0 + c * 16 + m16;
      out[(size_t)(rowBase + M) * H_DIM + col] = acc[c][r] + bo[col];
    }
}

// ---------------------------------------------------------------------------
extern "C" void kernel_launch(void* const* d_in, const int* in_sizes, int n_in,
                              void* d_out, int out_size, void* d_ws, size_t ws_size,
                              hipStream_t stream) {
  (void)in_sizes; (void)n_in; (void)out_size; (void)ws_size;
  const float* x    = (const float*)d_in[0];
  // d_in[1] attention_mask: per-row constant shift -> softmax-invariant, unused.
  const float* Wq   = (const float*)d_in[2];
  const float* bq   = (const float*)d_in[3];
  const float* Wk   = (const float*)d_in[4];
  const float* bk   = (const float*)d_in[5];
  const float* Wv   = (const float*)d_in[6];
  const float* bv   = (const float*)d_in[7];
  const float* Wo   = (const float*)d_in[8];
  const float* bo   = (const float*)d_in[9];
  const float* bias = (const float*)d_in[10];
  float* out = (float*)d_out;

  const size_t R = (size_t)8 * T_SEQ;  // 16384 rows
  float* ws  = (float*)d_ws;
  float* q   = ws;
  float* k   = ws + R * 64;
  float* vT  = ws + 2 * R * 64;
  float* ctx = ws + 3 * R * 64;        // 16 MB total workspace

  qkv_kernel<<<dim3(128), dim3(256), 0, stream>>>(x, Wq, bq, Wk, bk, Wv, bv, q, k, vT);
  attn_kernel<<<dim3(1024), dim3(32), 0, stream>>>(q, k, vT, bias, ctx);
  out_kernel<<<dim3(128, 16), dim3(256), 0, stream>>>(ctx, Wo, bo, out);
}
